// SAKEFlowModel_33071248179752
// MI455X (gfx1250) — matrix operand order repaired and checked by reading, not compiled, not run on hardware
//
#include <hip/hip_runtime.h>
#include <hip/hip_bf16.h>
#include <math.h>

// ---------------------------------------------------------------------------
// SAKE flow model for MI455X (gfx1250, wave32, WMMA)
// B=4, N=128, F=64, H=4, K=50, C=256, DEPTH=2, STEPS=3
// ---------------------------------------------------------------------------

#define NB   128   // nodes per batch
#define FD   64    // hidden dim
#define KRBF 50
#define CD   256   // F*H

typedef __attribute__((ext_vector_type(16))) _Float16 v16h;
typedef __attribute__((ext_vector_type(8)))  _Float16 v8h;
typedef __attribute__((ext_vector_type(4)))  _Float16 v4h;
typedef __attribute__((ext_vector_type(8)))  float    v8f;

// ---- WMMA helpers (CDNA5 V_WMMA_F32_16X16X32_F16, wave32) ------------------

__device__ __forceinline__ v8f wmma32(v16h a, v16h b, v8f c) {
  // (neg_a, A, neg_b, B, c_mod, C, reuse_a, reuse_b)
  return __builtin_amdgcn_wmma_f32_16x16x32_f16(false, a, false, b, (short)0, c,
                                                false, false);
}

// A fragment (16x32 f16) from LDS rows (row stride FD).
// ISA layout: lanes0-15 -> M=lane, K {0..7,16..23}; lanes16-31 -> K {8..15,24..31}
// => per lane: two contiguous 8-halfword runs -> 2x ds_load_b128
__device__ __forceinline__ v16h frag_a_rows(const _Float16* tile, int koff, int lane) {
  int m   = lane & 15;
  int kh8 = (lane >> 4) << 3;
  const _Float16* r0 = tile + m * FD + koff + kh8;
  v8h lo = *(const v8h*)(r0);
  v8h hi = *(const v8h*)(r0 + 16);
  return __builtin_shufflevector(lo, hi, 0, 1, 2, 3, 4, 5, 6, 7,
                                 8, 9, 10, 11, 12, 13, 14, 15);
}

// A fragment with every row equal to a broadcast vector (h_i block of hh)
__device__ __forceinline__ v16h frag_a_bcast(const _Float16* vec, int koff, int lane) {
  int kh8 = (lane >> 4) << 3;
  const _Float16* r0 = vec + koff + kh8;
  v8h lo = *(const v8h*)(r0);
  v8h hi = *(const v8h*)(r0 + 16);
  return __builtin_shufflevector(lo, hi, 0, 1, 2, 3, 4, 5, 6, 7,
                                 8, 9, 10, 11, 12, 13, 14, 15);
}

// B fragment from pre-packed f16 weights: fragment (nt,kc) is 32 lanes x 16 f16
// contiguous; per lane one 32B load (2x global_load_b128), zero-padding baked in.
__device__ __forceinline__ v16h frag_b_pk(const _Float16* __restrict__ P, int KC,
                                          int kc, int nt, int lane) {
  return *(const v16h*)(P + (((nt * KC + kc) << 9) + (lane << 4)));
}

__device__ __forceinline__ float silu_f(float x) { return x / (1.0f + __expf(-x)); }

// ---------------------------------------------------------------------------
// Pack a f32 weight matrix W[K][N] into WMMA-B fragment order, f16, padded to
// KC*32 x NT*16.  out[((nt*KC+kc)*32 + lane)*16 + s] = W[kc*32+k8+(s&7)+16*(s>>3)][nt*16+n]
__global__ __launch_bounds__(256) void pack_b_kernel(const float* __restrict__ W,
                                                     _Float16* __restrict__ out,
                                                     int K, int N, int KC, int NT) {
  int idx = blockIdx.x * 256 + threadIdx.x;
  int total = (KC * NT) << 9;
  if (idx >= total) return;
  int frag = idx >> 9;
  int rem  = idx & 511;
  int lane = rem >> 4;
  int s    = rem & 15;
  int nt = frag / KC, kc = frag - nt * KC;
  int k = kc * 32 + ((lane >> 4) << 3) + (s & 7) + ((s >> 3) << 4);
  int n = nt * 16 + (lane & 15);
  out[idx] = (k < K && n < N) ? (_Float16)W[k * N + n] : (_Float16)0.0f;
}

// ---------------------------------------------------------------------------
__global__ void init_out_kernel(const float* __restrict__ x, const float* __restrict__ v,
                                float* __restrict__ out) {
  int i = blockIdx.x * blockDim.x + threadIdx.x;
  if (i < 4 * NB * 3) { out[i] = x[i]; out[4 * NB * 3 + i] = v[i]; }
  if (i < 4) out[2 * 4 * NB * 3 + i] = 0.0f;
}

__global__ void copy_kernel(const float* __restrict__ s, float* __restrict__ d, int n) {
  int i = blockIdx.x * blockDim.x + threadIdx.x;
  if (i < n) d[i] = s[i];
}

// ---------------------------------------------------------------------------
// embedding: h0 = lin(silu(lin(h_raw, emb1)), emb2); one block per node
__global__ __launch_bounds__(64) void embed_kernel(
    const float* __restrict__ h_in, const float* __restrict__ w1,
    const float* __restrict__ b1, const float* __restrict__ w2,
    const float* __restrict__ b2, float* __restrict__ h0) {
  __shared__ float t1[FD];
  int node = blockIdx.x;
  int f = threadIdx.x;
  const float* hr = h_in + node * 16;
  float a = b1[f];
#pragma unroll
  for (int k = 0; k < 16; ++k) a += hr[k] * w1[k * FD + f];
  t1[f] = silu_f(a);
  __syncthreads();
  float o = b2[f];
  for (int k = 0; k < FD; ++k) o += t1[k] * w2[k * FD + f];
  h0[node * FD + f] = o;
}

// ---------------------------------------------------------------------------
// One message-passing step for node (blockIdx.x = i, blockIdx.y = b).
// 256 threads = 8 waves; wave w owns edge tile j = 16w .. 16w+15.
__global__ __launch_bounds__(256) void edge_mp_kernel(
    const float* __restrict__ h_in, const float* __restrict__ pos,
    float* __restrict__ h_out,
    const _Float16* __restrict__ pw_in, const float* __restrict__ b_in,
    const float* __restrict__ rbf_mu, const float* __restrict__ rbf_be,
    const _Float16* __restrict__ pw_e1, const float* __restrict__ b_e1,
    const _Float16* __restrict__ pw_e2, const float* __restrict__ b_e2,
    const _Float16* __restrict__ pw_mix,
    const float* __restrict__ W_p1,  const float* __restrict__ b_p1,
    const float* __restrict__ W_p2,  const float* __restrict__ b_p2,
    const float* __restrict__ W_sem, const float* __restrict__ b_sem,
    const float* __restrict__ W_n1,  const float* __restrict__ b_n1,
    const float* __restrict__ W_n2,  const float* __restrict__ b_n2) {
  __shared__ __align__(32) _Float16 sh_h[NB][FD];   // h_j f16; later he
  __shared__ __align__(32) _Float16 sh_ex[NB][FD];  // z extra block; later t
  __shared__ __align__(32) _Float16 sh_hif[FD];
  __shared__ float    sh_hi[FD];
  __shared__ float    sh_dxu[NB][3];
  __shared__ float    sh_dn[NB];
  __shared__ float    sh_att[NB][4];
  __shared__ float    sh_comb[CD][3];
  __shared__ float    sh_heagg[CD];
  __shared__ float    sh_cs[CD];
  __shared__ float    sh_v1[FD];
  __shared__ float    sh_v2[FD];

  const int ni  = blockIdx.x;
  const int bi  = blockIdx.y;
  const int tid = threadIdx.x;
  const int lane = tid & 31;
  const int wv   = tid >> 5;

  __builtin_prefetch(pw_e1, 0, 1);
  __builtin_prefetch(pw_mix, 0, 1);

  const float* hb = h_in + (size_t)(bi * NB) * FD;
  // ---- phase 0: stage h in f16 (vectorized), zero scratch, geometry ----
  {
    const float4* hb4 = (const float4*)hb;
    _Float16* dsth = &sh_h[0][0];
    for (int idx = tid; idx < NB * FD / 4; idx += 256) {
      float4 q = hb4[idx];
      v4h hv = {(_Float16)q.x, (_Float16)q.y, (_Float16)q.z, (_Float16)q.w};
      *(v4h*)(dsth + idx * 4) = hv;
    }
    v8h z8 = {};
    _Float16* dste = &sh_ex[0][0];
    for (int idx = tid; idx < NB * FD / 8; idx += 256)
      *(v8h*)(dste + idx * 8) = z8;
  }
  for (int idx = tid; idx < CD * 3; idx += 256) (&sh_comb[0][0])[idx] = 0.0f;
  if (tid < FD) {
    float hv = hb[ni * FD + tid];
    sh_hi[tid]  = hv;
    sh_hif[tid] = (_Float16)hv;
  }
  if (tid < NB) {
    const float* pb = pos + (size_t)(bi * NB) * 3;
    float d0 = pb[ni * 3]     - pb[tid * 3];
    float d1 = pb[ni * 3 + 1] - pb[tid * 3 + 1];
    float d2 = pb[ni * 3 + 2] - pb[tid * 3 + 2];
    float dn = sqrtf(d0 * d0 + d1 * d1 + d2 * d2 + 1e-10f);
    sh_dn[tid] = dn;
    float inv = 1.0f / (dn + 1e-5f);
    sh_dxu[tid][0] = d0 * inv; sh_dxu[tid][1] = d1 * inv; sh_dxu[tid][2] = d2 * inv;
  }
  __syncthreads();

  if (tid < NB) sh_ex[tid][KRBF] = (_Float16)sh_dn[tid];  // dn column of z

  const _Float16* atile = &sh_h[wv * 16][0];
  const _Float16* etile = &sh_ex[wv * 16][0];

  // ---- GEMM1: hk = hh @ W_in (K=128, N=50 pad 64) -> sh_ex (smear*hk) ----
  {
    v8f acc[4] = {};
#pragma unroll
    for (int kc = 0; kc < 4; ++kc) {
      v16h a = (kc < 2) ? frag_a_bcast(sh_hif, kc * 32, lane)
                        : frag_a_rows(atile, (kc - 2) * 32, lane);
#pragma unroll
      for (int nt = 0; nt < 4; ++nt)
        acc[nt] = wmma32(a, frag_b_pk(pw_in, 4, kc, nt, lane), acc[nt]);
    }
#pragma unroll
    for (int nt = 0; nt < 4; ++nt) {
      int col = nt * 16 + (lane & 15);
      if (col < KRBF) {
        float mu = rbf_mu[col], be = rbf_be[col], bv = b_in[col];
#pragma unroll
        for (int r = 0; r < 8; ++r) {
          int j = wv * 16 + r + ((lane >> 4) << 3);
          float de = __expf(-sh_dn[j]) - mu;
          float smear = __expf(-be * de * de);
          sh_ex[j][col] = (_Float16)(smear * (acc[nt][r] + bv));
        }
      }
    }
  }
  __syncthreads();  // dn column + smear*hk visible to all readers of sh_ex

  // ---- GEMM2: t = silu(z @ W_e1 + b), K = 179 pad 192 ----
  {
    v8f acc[4] = {};
#pragma unroll
    for (int kc = 0; kc < 6; ++kc) {
      v16h a = (kc < 2) ? frag_a_bcast(sh_hif, kc * 32, lane)
             : (kc < 4) ? frag_a_rows(atile, (kc - 2) * 32, lane)
                        : frag_a_rows(etile, (kc - 4) * 32, lane);
#pragma unroll
      for (int nt = 0; nt < 4; ++nt)
        acc[nt] = wmma32(a, frag_b_pk(pw_e1, 6, kc, nt, lane), acc[nt]);
    }
#pragma unroll
    for (int nt = 0; nt < 4; ++nt) {
      int col = nt * 16 + (lane & 15);
      float bv = b_e1[col];
#pragma unroll
      for (int r = 0; r < 8; ++r) {
        int j = wv * 16 + r + ((lane >> 4) << 3);
        sh_ex[j][col] = (_Float16)silu_f(acc[nt][r] + bv);  // overwrite ex with t
      }
    }
  }

  // ---- GEMM3: he = t @ W_e2 + b -> sh_h (reuse) ----
  {
    v8f acc[4] = {};
#pragma unroll
    for (int kc = 0; kc < 2; ++kc) {
      v16h a = frag_a_rows(etile, kc * 32, lane);
#pragma unroll
      for (int nt = 0; nt < 4; ++nt)
        acc[nt] = wmma32(a, frag_b_pk(pw_e2, 2, kc, nt, lane), acc[nt]);
    }
#pragma unroll
    for (int nt = 0; nt < 4; ++nt) {
      int col = nt * 16 + (lane & 15);
      float bv = b_e2[col];
#pragma unroll
      for (int r = 0; r < 8; ++r) {
        int j = wv * 16 + r + ((lane >> 4) << 3);
        sh_h[j][col] = (_Float16)(acc[nt][r] + bv);
      }
    }
  }
  __syncthreads();  // he visible to all

  // ---- xmix GEMM + spatial comb accumulation (coeff = tanh(he@W_mix)) ----
  for (int nt = 0; nt < 16; ++nt) {
    v8f acc = {};
#pragma unroll
    for (int kc = 0; kc < 2; ++kc) {
      v16h a = frag_a_rows(atile, kc * 32, lane);
      acc = wmma32(a, frag_b_pk(pw_mix, 2, kc, nt, lane), acc);
    }
    float p0 = 0.f, p1 = 0.f, p2 = 0.f;
#pragma unroll
    for (int r = 0; r < 8; ++r) {
      int j = wv * 16 + r + ((lane >> 4) << 3);
      float cf = tanhf(acc[r]);
      p0 += cf * sh_dxu[j][0];
      p1 += cf * sh_dxu[j][1];
      p2 += cf * sh_dxu[j][2];
    }
    p0 += __shfl_xor(p0, 16);
    p1 += __shfl_xor(p1, 16);
    p2 += __shfl_xor(p2, 16);
    if (lane < 16) {
      int c = nt * 16 + lane;
      atomicAdd(&sh_comb[c][0], p0);
      atomicAdd(&sh_comb[c][1], p1);
      atomicAdd(&sh_comb[c][2], p2);
    }
  }

  // ---- semantic attention logits (celu alpha=2, self masked) ----
  if (tid < NB) {
    int j = tid;
    for (int hh = 0; hh < 4; ++hh) {
      float a = b_sem[hh];
      for (int f = 0; f < FD; ++f) a += (float)sh_h[j][f] * W_sem[f * 4 + hh];
      a = (a >= 0.f) ? a : 2.f * (__expf(0.5f * a) - 1.f);
      if (j == ni) a -= 1e5f;
      sh_att[j][hh] = a;
    }
  }
  __syncthreads();

  // ---- softmax over j per head ----
  if (tid < 4) {
    int hh = tid;
    float mx = -3.4e38f;
    for (int j = 0; j < NB; ++j) mx = fmaxf(mx, sh_att[j][hh]);
    float s = 0.f;
    for (int j = 0; j < NB; ++j) {
      float e = __expf(sh_att[j][hh] - mx);
      sh_att[j][hh] = e;
      s += e;
    }
    float inv = 1.f / s;
    for (int j = 0; j < NB; ++j) sh_att[j][hh] *= inv;
  }
  __syncthreads();

  // ---- h_e aggregation + comb^2 ----
  {
    int f = tid >> 2, hh = tid & 3;
    float s = 0.f;
    for (int j = 0; j < NB; ++j) s += (float)sh_h[j][f] * sh_att[j][hh];
    sh_heagg[f * 4 + hh] = s;
    float c0 = sh_comb[tid][0] * (1.f / NB);
    float c1 = sh_comb[tid][1] * (1.f / NB);
    float c2 = sh_comb[tid][2] * (1.f / NB);
    sh_cs[tid] = c0 * c0 + c1 * c1 + c2 * c2;
  }
  __syncthreads();

  // ---- node MLPs (small, VALU) ----
  if (tid < FD) {
    float a = b_p1[tid];
    for (int c = 0; c < CD; ++c) a += sh_cs[c] * W_p1[c * FD + tid];
    sh_v1[tid] = silu_f(a);
  }
  __syncthreads();
  if (tid < FD) {
    float a = b_p2[tid];
    for (int k = 0; k < FD; ++k) a += sh_v1[k] * W_p2[k * FD + tid];
    sh_v2[tid] = silu_f(a);  // hc
  }
  __syncthreads();
  if (tid < FD) {
    float a = b_n1[tid];
    for (int k = 0; k < FD; ++k) a += sh_hi[k] * W_n1[k * FD + tid];
    for (int c = 0; c < CD; ++c) a += sh_heagg[c] * W_n1[(FD + c) * FD + tid];
    for (int k = 0; k < FD; ++k) a += sh_v2[k] * W_n1[(FD + CD + k) * FD + tid];
    sh_cs[tid] = silu_f(a);
  }
  __syncthreads();
  if (tid < FD) {
    float a = b_n2[tid];
    for (int k = 0; k < FD; ++k) a += sh_cs[k] * W_n2[k * FD + tid];
    h_out[(size_t)(bi * NB + ni) * FD + tid] = sh_hi[tid] + silu_f(a);
  }
}

// ---------------------------------------------------------------------------
// Flow tail: coordinate update (WMMA over j) + velocity scale + log-det
__global__ __launch_bounds__(256) void flow_tail_kernel(
    const float* __restrict__ h, const float* __restrict__ pos,
    const float* __restrict__ vel, float* __restrict__ pos_out,
    float* __restrict__ vel_out, float* __restrict__ sld,
    const _Float16* __restrict__ pw_c1, const float* __restrict__ b_c1,
    const float* __restrict__ W_c2, const float* __restrict__ W_v1,
    const float* __restrict__ b_v1, const float* __restrict__ W_v2) {
  __shared__ __align__(32) _Float16 sh_h[NB][FD];
  __shared__ __align__(32) _Float16 sh_hif[FD];
  __shared__ __align__(32) _Float16 sh_t[NB][FD];
  __shared__ float    sh_hi[FD];
  __shared__ float    sh_dx[NB][3];
  __shared__ float    sh_c[NB];
  __shared__ float    sh_m1[FD];
  __shared__ float    sh_dv[3];
  __shared__ float    sh_m;

  const int ni = blockIdx.x, bi = blockIdx.y;
  const int tid = threadIdx.x;
  const int lane = tid & 31, wv = tid >> 5;

  const float* hb = h + (size_t)(bi * NB) * FD;
  {
    const float4* hb4 = (const float4*)hb;
    _Float16* dsth = &sh_h[0][0];
    for (int idx = tid; idx < NB * FD / 4; idx += 256) {
      float4 q = hb4[idx];
      v4h hv = {(_Float16)q.x, (_Float16)q.y, (_Float16)q.z, (_Float16)q.w};
      *(v4h*)(dsth + idx * 4) = hv;
    }
  }
  if (tid < FD) {
    float hv = hb[ni * FD + tid];
    sh_hi[tid]  = hv;
    sh_hif[tid] = (_Float16)hv;
  }
  if (tid < NB) {
    const float* pb = pos + (size_t)(bi * NB) * 3;
    sh_dx[tid][0] = pb[ni * 3]     - pb[tid * 3];
    sh_dx[tid][1] = pb[ni * 3 + 1] - pb[tid * 3 + 1];
    sh_dx[tid][2] = pb[ni * 3 + 2] - pb[tid * 3 + 2];
  }
  __syncthreads();

  // t = silu(hh @ W_c1 + b): K=128 (h_i bcast | h_j), N=64
  const _Float16* atile = &sh_h[wv * 16][0];
  {
    v8f acc[4] = {};
#pragma unroll
    for (int kc = 0; kc < 4; ++kc) {
      v16h a = (kc < 2) ? frag_a_bcast(sh_hif, kc * 32, lane)
                        : frag_a_rows(atile, (kc - 2) * 32, lane);
#pragma unroll
      for (int nt = 0; nt < 4; ++nt)
        acc[nt] = wmma32(a, frag_b_pk(pw_c1, 4, kc, nt, lane), acc[nt]);
    }
#pragma unroll
    for (int nt = 0; nt < 4; ++nt) {
      int col = nt * 16 + (lane & 15);
      float bv = b_c1[col];
#pragma unroll
      for (int r = 0; r < 8; ++r) {
        int j = wv * 16 + r + ((lane >> 4) << 3);
        sh_t[j][col] = (_Float16)silu_f(acc[nt][r] + bv);
      }
    }
  }
  // velocity gate MLP layer 1 (independent of GEMM)
  if (tid < FD) {
    float a = b_v1[tid];
    for (int k = 0; k < FD; ++k) a += sh_hi[k] * W_v1[k * FD + tid];
    sh_m1[tid] = silu_f(a);
  }
  __syncthreads();

  if (tid < NB) {
    float s = 0.f;
    for (int f = 0; f < FD; ++f) s += (float)sh_t[tid][f] * W_c2[f];
    sh_c[tid] = s;
  }
  if (tid == 0) {
    float m = 0.f;
    for (int k = 0; k < FD; ++k) m += sh_m1[k] * W_v2[k];
    sh_m = m;
  }
  __syncthreads();

  if (tid < 3) {
    float s = 0.f;
    for (int j = 0; j < NB; ++j) s += sh_c[j] * sh_dx[j][tid];
    sh_dv[tid] = s * (1.f / NB);
  }
  __syncthreads();

  if (tid == 0) {
    float m = sh_m;
    float em = __expf(m);
    int base = (bi * NB + ni) * 3;
#pragma unroll
    for (int d = 0; d < 3; ++d) {
      float vn = em * vel[base + d] + sh_dv[d];
      vel_out[base + d] = vn;
      pos_out[base + d] = pos[base + d] + vn;
    }
    atomicAdd(&sld[bi], 3.0f * m);
  }
}

// ---------------------------------------------------------------------------
__global__ __launch_bounds__(128) void center_kernel(const float* __restrict__ src,
                                                     float* __restrict__ dst) {
  __shared__ float s[3];
  int bi = blockIdx.x, j = threadIdx.x;
  if (j < 3) s[j] = 0.f;
  __syncthreads();
  int base = (bi * NB + j) * 3;
  float v0 = src[base], v1 = src[base + 1], v2 = src[base + 2];
  atomicAdd(&s[0], v0);
  atomicAdd(&s[1], v1);
  atomicAdd(&s[2], v2);
  __syncthreads();
  dst[base]     = v0 - s[0] * (1.f / NB);
  dst[base + 1] = v1 - s[1] * (1.f / NB);
  dst[base + 2] = v2 - s[2] * (1.f / NB);
}

// ---------------------------------------------------------------------------
// packed-weight region layout per layer (f16 elements):
//   win:   0     .. 8191    (KC=4,  NT=4)
//   we1:   8192  .. 20479   (KC=6,  NT=4)
//   we2:   20480 .. 24575   (KC=2,  NT=4)
//   wmix:  24576 .. 40959   (KC=2,  NT=16)
//   wc1:   40960 .. 49151   (KC=4,  NT=4)
#define PK_LAYER 49152

extern "C" void kernel_launch(void* const* d_in, const int* in_sizes, int n_in,
                              void* d_out, int out_size, void* d_ws, size_t ws_size,
                              hipStream_t stream) {
  (void)in_sizes; (void)n_in; (void)out_size; (void)ws_size;
  const float* h_raw = (const float*)d_in[0];
  const float* x_in  = (const float*)d_in[1];
  const float* v_in  = (const float*)d_in[2];
  const float* emb1_w = (const float*)d_in[3];
  const float* emb1_b = (const float*)d_in[4];
  const float* emb2_w = (const float*)d_in[5];
  const float* emb2_b = (const float*)d_in[6];
  const float* edge_in_w = (const float*)d_in[7];
  const float* edge_in_b = (const float*)d_in[8];
  const float* rbf_means = (const float*)d_in[9];
  const float* rbf_betas = (const float*)d_in[10];
  const float* eo1_w = (const float*)d_in[11];
  const float* eo1_b = (const float*)d_in[12];
  const float* eo2_w = (const float*)d_in[13];
  const float* eo2_b = (const float*)d_in[14];
  const float* xmix_w = (const float*)d_in[15];
  const float* post1_w = (const float*)d_in[16];
  const float* post1_b = (const float*)d_in[17];
  const float* post2_w = (const float*)d_in[18];
  const float* post2_b = (const float*)d_in[19];
  const float* sem_w = (const float*)d_in[20];
  const float* sem_b = (const float*)d_in[21];
  const float* node1_w = (const float*)d_in[22];
  const float* node1_b = (const float*)d_in[23];
  const float* node2_w = (const float*)d_in[24];
  const float* node2_b = (const float*)d_in[25];
  const float* vel1_w = (const float*)d_in[26];
  const float* vel1_b = (const float*)d_in[27];
  const float* vel2_w = (const float*)d_in[28];
  const float* coord1_w = (const float*)d_in[29];
  const float* coord1_b = (const float*)d_in[30];
  const float* coord2_w = (const float*)d_in[31];

  float* out  = (float*)d_out;
  float* xbuf = out;                 // [B,N,3]
  float* vbuf = out + 4 * NB * 3;    // [B,N,3]
  float* sld  = out + 8 * NB * 3;    // [B]

  float* ws = (float*)d_ws;
  float* h0 = ws;                    // [B,N,F] = 32768 floats
  float* hA = ws + 32768;
  float* hB = ws + 65536;
  float* xt = ws + 98304;            // pos temp [B,N,3]
  float* vt = xt + 4 * NB * 3;       // vel temp
  _Float16* pack = (_Float16*)(ws + 101376);  // 32B-aligned packed weights

  // ---- pack all GEMM weights (f32 -> f16 fragment layout), every call ----
  for (int l = 0; l < 4; ++l) {
    _Float16* pl = pack + l * PK_LAYER;
    pack_b_kernel<<<dim3(32), 256, 0, stream>>>(edge_in_w + l * 2 * FD * KRBF,
                                                pl + 0, 2 * FD, KRBF, 4, 4);
    pack_b_kernel<<<dim3(48), 256, 0, stream>>>(eo1_w + l * (2 * FD + KRBF + 1) * FD,
                                                pl + 8192, 2 * FD + KRBF + 1, FD, 6, 4);
    pack_b_kernel<<<dim3(16), 256, 0, stream>>>(eo2_w + l * FD * FD,
                                                pl + 20480, FD, FD, 2, 4);
    pack_b_kernel<<<dim3(64), 256, 0, stream>>>(xmix_w + l * FD * CD,
                                                pl + 24576, FD, CD, 2, 16);
    pack_b_kernel<<<dim3(32), 256, 0, stream>>>(coord1_w + l * 2 * FD * FD,
                                                pl + 40960, 2 * FD, FD, 4, 4);
  }

  init_out_kernel<<<dim3(6), 256, 0, stream>>>(x_in, v_in, out);
  embed_kernel<<<dim3(4 * NB), 64, 0, stream>>>(h_raw, emb1_w, emb1_b, emb2_w,
                                                emb2_b, h0);

  for (int d = 0; d < 2; ++d) {
    for (int ph = 0; ph < 2; ++ph) {
      int l = 2 * d + ph;
      _Float16* pl = pack + l * PK_LAYER;
      // h is re-initialized from the embedding for every flow layer
      copy_kernel<<<dim3(128), 256, 0, stream>>>(h0, hA, 4 * NB * FD);
      float* posb = ph ? vbuf : xbuf;  // role swap on odd layers
      float* velb = ph ? xbuf : vbuf;
      float* cur = hA;
      float* nxt = hB;
      for (int s = 0; s < 3; ++s) {
        edge_mp_kernel<<<dim3(NB, 4), 256, 0, stream>>>(
            cur, posb, nxt,
            pl + 0, edge_in_b + l * KRBF,
            rbf_means + l * KRBF, rbf_betas + l * KRBF,
            pl + 8192, eo1_b + l * FD,
            pl + 20480, eo2_b + l * FD,
            pl + 24576,
            post1_w + l * CD * FD, post1_b + l * FD,
            post2_w + l * FD * FD, post2_b + l * FD,
            sem_w + l * FD * 4, sem_b + l * 4,
            node1_w + l * (FD + CD + FD) * FD, node1_b + l * FD,
            node2_w + l * FD * FD, node2_b + l * FD);
        float* tmp = cur; cur = nxt; nxt = tmp;
      }
      flow_tail_kernel<<<dim3(NB, 4), 256, 0, stream>>>(
          cur, posb, velb, xt, vt, sld,
          pl + 40960, coord1_b + l * FD, coord2_w + l * FD,
          vel1_w + l * FD * FD, vel1_b + l * FD, vel2_w + l * FD);
      center_kernel<<<dim3(4), 128, 0, stream>>>(xt, posb);
      center_kernel<<<dim3(4), 128, 0, stream>>>(vt, velb);
    }
  }
}
